// MultiHeadedAttention_63376537420578
// MI455X (gfx1250) — matrix operand order repaired and checked by reading, not compiled, not run on hardware
//
#include <hip/hip_runtime.h>

typedef __attribute__((ext_vector_type(16))) __bf16 v16bf;
typedef __attribute__((ext_vector_type(8)))  float  v8f;

#define WMMA_BF16(A, B, C) \
  __builtin_amdgcn_wmma_f32_16x16x32_bf16(false, (A), false, (B), (short)0, (C), false, false)

// ---------------------------------------------------------------------------
// GEMM: Y[4096,1024] = X[4096,1024] @ W[1024,1024] + bias
// Template params resolve all hot-loop branches at compile time.
//   MODE 0: bf16, head layout  out[((b*16+h)*2048 + s)*64 + d]   (Q, K)
//   MODE 1: bf16, transposed   out[((b*16+h)*64 + d)*2048 + s]   (V^T)
//   MODE 2: fp32, flat         out[m*1024 + n]                   (final)
// Block tile 128x128, 256 threads = 8 waves, each wave 32x64 output.
// Software-pipelined: global loads for tile k+1 overlap WMMAs for tile k.
// ---------------------------------------------------------------------------
template <int XBF16, int MODE>
__global__ __launch_bounds__(256)
void gemm_wmma_kernel(const void* __restrict__ Xv,
                      const float* __restrict__ W,
                      const float* __restrict__ bias,
                      void* __restrict__ outv)
{
  const int N = 1024, K = 1024;
  __shared__ __bf16 lA[128 * 32];   // [m][k]
  __shared__ __bf16 lB[128 * 32];   // [n][k] (transposed tile of W)

  const int tid   = threadIdx.x;
  const int m0    = blockIdx.x * 128;
  const int n0    = blockIdx.y * 128;
  const int wave  = tid >> 5;
  const int lane  = tid & 31;
  const int lh    = lane >> 4;      // lane-half
  const int ln    = lane & 15;
  const int mslab = (wave & 3) * 32;
  const int nslab = (wave >> 2) * 64;

  v8f acc[2][4];
#pragma unroll
  for (int a = 0; a < 2; ++a)
#pragma unroll
    for (int j = 0; j < 4; ++j)
#pragma unroll
      for (int i = 0; i < 8; ++i) acc[a][j][i] = 0.0f;

  // staging indices: A 128x32 (16 elems/thread), B 32x128 (2k x 8n /thread)
  const int ar = tid >> 1;           // 0..127
  const int ac = (tid & 1) * 16;     // 0 or 16
  const int kp = (tid & 15) * 2;     // even k pair base 0..30
  const int nb = (tid >> 4) * 8;     // 0..120

  // pipelined staging registers
  float aReg[16];
  uint4 aRegB[2];
  float bReg0[8], bReg1[8];

  auto loadGlobal = [&](int kb) {
    if (XBF16) {
      const __bf16* X = (const __bf16*)Xv;
      const uint4* p = (const uint4*)(X + (size_t)(m0 + ar) * K + kb + ac);
      aRegB[0] = p[0];
      aRegB[1] = p[1];
    } else {
      const float* X = (const float*)Xv;
      const float* s = X + (size_t)(m0 + ar) * K + kb + ac;
#pragma unroll
      for (int j = 0; j < 16; ++j) aReg[j] = s[j];
    }
    const float* w0 = W + (size_t)(kb + kp) * N + n0 + nb;
    const float* w1 = w0 + N;
#pragma unroll
    for (int j = 0; j < 8; ++j) { bReg0[j] = w0[j]; bReg1[j] = w1[j]; }
  };

  auto stageLDS = [&]() {
    if (XBF16) {
      *(uint4*)(&lA[ar * 32 + ac])     = aRegB[0];
      *(uint4*)(&lA[ar * 32 + ac + 8]) = aRegB[1];
    } else {
#pragma unroll
      for (int j = 0; j < 16; ++j) lA[ar * 32 + ac + j] = (__bf16)aReg[j];
    }
#pragma unroll
    for (int j = 0; j < 8; ++j) {
      union { __bf16 h[2]; unsigned int u; } pk;
      pk.h[0] = (__bf16)bReg0[j];
      pk.h[1] = (__bf16)bReg1[j];
      *(unsigned int*)(&lB[(nb + j) * 32 + kp]) = pk.u;
    }
  };

  loadGlobal(0);
  for (int kb = 0; kb < K; kb += 32) {
    stageLDS();
    __syncthreads();
    if (kb + 32 < K) loadGlobal(kb + 32);   // overlap with compute below

    v16bf afr[2];
#pragma unroll
    for (int a = 0; a < 2; ++a) {
      const __bf16* row = &lA[(mslab + a * 16 + ln) * 32];
      ((uint4*)&afr[a])[0] = *(const uint4*)(row + 16 * lh);
      ((uint4*)&afr[a])[1] = *(const uint4*)(row + 16 * lh + 8);
    }
#pragma unroll
    for (int j = 0; j < 4; ++j) {
      v16bf bfr = *(const v16bf*)(&lB[(nslab + j * 16 + ln) * 32 + 16 * lh]);
      acc[0][j] = WMMA_BF16(afr[0], bfr, acc[0][j]);
      acc[1][j] = WMMA_BF16(afr[1], bfr, acc[1][j]);
    }
    __syncthreads();
  }

  // ---- epilogue: bias + layout-specific store ----
#pragma unroll
  for (int a = 0; a < 2; ++a) {
#pragma unroll
    for (int j = 0; j < 4; ++j) {
      const int n = n0 + nslab + j * 16 + ln;
      const float bval = bias[n];
#pragma unroll
      for (int i = 0; i < 8; ++i) {
        const int m = m0 + mslab + a * 16 + i + 8 * lh;
        const float v = acc[a][j][i] + bval;
        if (MODE == 2) {
          ((float*)outv)[(size_t)m * 1024 + n] = v;
        } else {
          const int bb = m >> 11, s = m & 2047;
          const int h  = n >> 6,  d = n & 63;
          __bf16* O = (__bf16*)outv;
          if (MODE == 0)
            O[(((size_t)(bb * 16 + h) * 2048) + s) * 64 + d] = (__bf16)v;
          else
            O[(((size_t)(bb * 16 + h) * 64) + d) * 2048 + s] = (__bf16)v;
        }
      }
    }
  }
}

// ---------------------------------------------------------------------------
// Flash attention: one wave handles 16 query rows, streaming 32-key tiles.
// Q,K: bf16 [B,H,S,64]; V: bf16 [B,H,64,S] (transposed); out: bf16 [B,S,1024].
// ---------------------------------------------------------------------------
__global__ __launch_bounds__(128)
void attn_wmma_kernel(const __bf16* __restrict__ Qb, const __bf16* __restrict__ Kb,
                      const __bf16* __restrict__ Vt, const int* __restrict__ mask,
                      __bf16* __restrict__ Ob)
{
  const int S = 2048, DK = 64;
  __shared__ __bf16 lP[4][16 * 32];   // per-wave P buffer (C-layout -> A-layout)

  const int wave = threadIdx.x >> 5;
  const int lane = threadIdx.x & 31;
  const int lh   = lane >> 4;
  const int ln   = lane & 15;
  const int gw   = blockIdx.x * 4 + wave;   // 0..4095
  const int qt   = gw & 127;                // query tile in sequence
  const int bh   = gw >> 7;                 // 0..31 (b*16+h)
  const int b    = bh >> 4;
  const int h    = bh & 15;
  const int q0   = qt * 16;

  const __bf16* Qp = Qb + (size_t)bh * S * DK;
  const __bf16* Kp = Kb + (size_t)bh * S * DK;
  const __bf16* Vp = Vt + (size_t)bh * DK * S;
  const int*    Mp = mask + (size_t)b * S * S;
  __bf16* pbuf = &lP[wave][0];

  // Q fragments: 2 chunks of contraction dim (d = 0..31, 32..63)
  v16bf qf[2];
#pragma unroll
  for (int c = 0; c < 2; ++c) {
    const __bf16* row = Qp + (size_t)(q0 + ln) * DK + c * 32;
    ((uint4*)&qf[c])[0] = *(const uint4*)(row + 16 * lh);
    ((uint4*)&qf[c])[1] = *(const uint4*)(row + 16 * lh + 8);
  }

  v8f o[4];
#pragma unroll
  for (int j = 0; j < 4; ++j)
#pragma unroll
    for (int i = 0; i < 8; ++i) o[j][i] = 0.0f;

  float mrun[8], srun[8];
#pragma unroll
  for (int i = 0; i < 8; ++i) { mrun[i] = -3.0e38f; srun[i] = 0.0f; }

  const float scale = 0.125f;  // 1/sqrt(64)

  for (int k0 = 0; k0 < S; k0 += 32) {
    // ---- scores: S_blk = Q @ K^T (two 16x16 tiles) ----
    v8f c0, c1;
#pragma unroll
    for (int i = 0; i < 8; ++i) { c0[i] = 0.0f; c1[i] = 0.0f; }
#pragma unroll
    for (int c = 0; c < 2; ++c) {
      v16bf b0 = *(const v16bf*)(Kp + (size_t)(k0 + ln) * DK + c * 32 + 16 * lh);
      c0 = WMMA_BF16(qf[c], b0, c0);
      v16bf b1 = *(const v16bf*)(Kp + (size_t)(k0 + 16 + ln) * DK + c * 32 + 16 * lh);
      c1 = WMMA_BF16(qf[c], b1, c1);
    }

    // ---- scale + mask ----
    float s0[8], s1[8];
#pragma unroll
    for (int i = 0; i < 8; ++i) {
      const int qrow = q0 + i + 8 * lh;
      const int* mrow = Mp + (size_t)qrow * S + k0;
      s0[i] = (mrow[ln]      != 0) ? c0[i] * scale : -1.0e9f;
      s1[i] = (mrow[16 + ln] != 0) ? c1[i] * scale : -1.0e9f;
    }

    // ---- tile row-max (butterfly within 16-lane half) ----
    float tm[8];
#pragma unroll
    for (int i = 0; i < 8; ++i) tm[i] = fmaxf(s0[i], s1[i]);
#pragma unroll
    for (int off = 1; off < 16; off <<= 1)
#pragma unroll
      for (int i = 0; i < 8; ++i) tm[i] = fmaxf(tm[i], __shfl_xor(tm[i], off, 32));

    // ---- online softmax update ----
    float al[8], rs[8];
#pragma unroll
    for (int i = 0; i < 8; ++i) {
      const float mnew = fmaxf(mrun[i], tm[i]);
      al[i] = __expf(mrun[i] - mnew);
      mrun[i] = mnew;
      s0[i] = __expf(s0[i] - mnew);
      s1[i] = __expf(s1[i] - mnew);
      rs[i] = s0[i] + s1[i];
    }
#pragma unroll
    for (int off = 1; off < 16; off <<= 1)
#pragma unroll
      for (int i = 0; i < 8; ++i) rs[i] += __shfl_xor(rs[i], off, 32);
#pragma unroll
    for (int i = 0; i < 8; ++i) srun[i] = srun[i] * al[i] + rs[i];
#pragma unroll
    for (int j = 0; j < 4; ++j)
#pragma unroll
      for (int i = 0; i < 8; ++i) o[j][i] *= al[i];

    // ---- P: C-layout -> LDS -> A-layout fragment ----
#pragma unroll
    for (int i = 0; i < 8; ++i) {
      const int m = i + 8 * lh;
      pbuf[m * 32 + ln]      = (__bf16)s0[i];
      pbuf[m * 32 + 16 + ln] = (__bf16)s1[i];
    }
    asm volatile("s_wait_dscnt 0x0" ::: "memory");
    v16bf pf;
    ((uint4*)&pf)[0] = *(const uint4*)(pbuf + ln * 32 + 16 * lh);
    ((uint4*)&pf)[1] = *(const uint4*)(pbuf + ln * 32 + 16 * lh + 8);

    // ---- O += P @ V_blk (V stored [d][t], contiguous in t) ----
#pragma unroll
    for (int j = 0; j < 4; ++j) {
      v16bf vf = *(const v16bf*)(Vp + (size_t)(j * 16 + ln) * S + k0 + 16 * lh);
      o[j] = WMMA_BF16(pf, vf, o[j]);
    }
  }

  // ---- normalize and store to [B,S,1024] bf16 ----
#pragma unroll
  for (int j = 0; j < 4; ++j) {
    const int d = j * 16 + ln;
#pragma unroll
    for (int i = 0; i < 8; ++i) {
      const int qrow = q0 + i + 8 * lh;
      const float v = o[j][i] / srun[i];
      Ob[((size_t)(b * 2048 + qrow)) * 1024 + h * 64 + d] = (__bf16)v;
    }
  }
}

// ---------------------------------------------------------------------------
extern "C" void kernel_launch(void* const* d_in, const int* in_sizes, int n_in,
                              void* d_out, int out_size, void* d_ws, size_t ws_size,
                              hipStream_t stream) {
  const float* q    = (const float*)d_in[0];
  const float* k    = (const float*)d_in[1];
  const float* v    = (const float*)d_in[2];
  const int*   mask = (const int*)d_in[3];
  const float* Wq   = (const float*)d_in[4];
  const float* bq   = (const float*)d_in[5];
  const float* Wk   = (const float*)d_in[6];
  const float* bk   = (const float*)d_in[7];
  const float* Wv   = (const float*)d_in[8];
  const float* bv   = (const float*)d_in[9];
  const float* Wo   = (const float*)d_in[10];
  const float* bo   = (const float*)d_in[11];

  const size_t ELEMS = (size_t)4096 * 1024;  // 4M bf16 elems = 8 MB each
  __bf16* qbuf  = (__bf16*)d_ws;
  __bf16* kbuf  = qbuf + ELEMS;
  __bf16* vtbuf = kbuf + ELEMS;
  __bf16* abuf  = vtbuf + ELEMS;

  dim3 gg(32, 8), gb(256);
  gemm_wmma_kernel<0, 0><<<gg, gb, 0, stream>>>(q, Wq, bq, qbuf);
  gemm_wmma_kernel<0, 0><<<gg, gb, 0, stream>>>(k, Wk, bk, kbuf);
  gemm_wmma_kernel<0, 1><<<gg, gb, 0, stream>>>(v, Wv, bv, vtbuf);

  attn_wmma_kernel<<<dim3(1024), dim3(128), 0, stream>>>(qbuf, kbuf, vtbuf, mask, abuf);

  gemm_wmma_kernel<1, 2><<<gg, gb, 0, stream>>>(abuf, Wo, bo, d_out);
}
